// NonLocalBlock_32272384262666
// MI455X (gfx1250) — matrix operand order repaired and checked by reading, not compiled, not run on hardware
//
#include <hip/hip_runtime.h>
#include <hip/hip_bf16.h>

typedef __attribute__((ext_vector_type(16))) _Float16 v16h;
typedef __attribute__((ext_vector_type(8)))  float    v8f;
typedef __attribute__((ext_vector_type(4)))  float    f4;
typedef __attribute__((ext_vector_type(4)))  int      v4i;

union HFrag { v16h v; _Float16 h[16]; };

// ---- optional CDNA5 async global->LDS path (guarded; falls back to sync copy) ---
#ifndef HAVE_ASYNC_LDS
#define HAVE_ASYNC_LDS 0
#endif
#if defined(__has_builtin)
#if __has_builtin(__builtin_amdgcn_global_load_async_to_lds_b128)
#undef HAVE_ASYNC_LDS
#define HAVE_ASYNC_LDS 1
#endif
#endif

#define AS_GLOBAL __attribute__((address_space(1)))
#define AS_LDS    __attribute__((address_space(3)))

__device__ __forceinline__ void stage16(void* lds_dst, const void* gsrc) {
#if HAVE_ASYNC_LDS
  __builtin_amdgcn_global_load_async_to_lds_b128(
      (AS_GLOBAL v4i*)gsrc, (AS_LDS v4i*)lds_dst, 0, 0);
#else
  *(f4*)lds_dst = *(const f4*)gsrc;
#endif
}

__device__ __forceinline__ void stage_wait() {
#if HAVE_ASYNC_LDS
#if __has_builtin(__builtin_amdgcn_s_wait_asynccnt)
  __builtin_amdgcn_s_wait_asynccnt(0);
#else
  asm volatile("s_wait_asynccnt 0" ::: "memory");
#endif
#endif
}

// ---------------- WMMA fragment helpers (wave32; A and B both use k = 16*hi + v) ----

__device__ __forceinline__ v8f wmma_f16(v16h a, v16h b, v8f c) {
  return __builtin_amdgcn_wmma_f32_16x16x32_f16(false, a, false, b, (short)0, c,
                                                false, false);
}

// A fragment 16x32 (MxK) from row-major LDS s[m][k], leading dim ld.
__device__ __forceinline__ v16h load_a_rowmajor(const _Float16* s, int ld, int m0, int k0) {
  HFrag f;
  const int lane = threadIdx.x & 31, hi = lane >> 4;
  const int m = m0 + (lane & 15);
#pragma unroll
  for (int v = 0; v < 16; ++v) {
    int k = k0 + v + 16 * hi;
    f.h[v] = s[m * ld + k];
  }
  return f.v;
}

// A fragment 16x32 where storage is s[k][m] (element (m,k) = s[k*ld+m]).
__device__ __forceinline__ v16h load_a_colmajor(const _Float16* s, int ld, int m0, int k0) {
  HFrag f;
  const int lane = threadIdx.x & 31, hi = lane >> 4;
  const int m = m0 + (lane & 15);
#pragma unroll
  for (int v = 0; v < 16; ++v) {
    int k = k0 + v + 16 * hi;
    f.h[v] = s[k * ld + m];
  }
  return f.v;
}

// B fragment 32x16 (KxN) from row-major s[k][n].
__device__ __forceinline__ v16h load_b_rowmajor(const _Float16* s, int ld, int k0, int n0) {
  HFrag f;
  const int lane = threadIdx.x & 31, hi = lane >> 4;
  const int n = n0 + (lane & 15);
#pragma unroll
  for (int v = 0; v < 16; ++v) {
    int k = k0 + v + 16 * hi;
    f.h[v] = s[k * ld + n];
  }
  return f.v;
}

// B fragment 32x16 where storage is s[n][k] (element (k,n) = s[n*ld+k]).
__device__ __forceinline__ v16h load_b_trans(const _Float16* s, int ld, int k0, int n0) {
  HFrag f;
  const int lane = threadIdx.x & 31, hi = lane >> 4;
  const int n = n0 + (lane & 15);
#pragma unroll
  for (int v = 0; v < 16; ++v) {
    int k = k0 + v + 16 * hi;
    f.h[v] = s[n * ld + k];
  }
  return f.v;
}

// ---------------- small utility kernels -----------------------------------------

__global__ void cvt_f32_f16(const float* __restrict__ src, _Float16* __restrict__ dst, int n) {
  int i = blockIdx.x * 256 + threadIdx.x;
  if (i < n) dst[i] = (_Float16)src[i];
}

// fold conv bias + inference BN into per-channel scale/shift
__global__ void bnprep(const float* __restrict__ g, const float* __restrict__ be,
                       const float* __restrict__ mu, const float* __restrict__ var,
                       const float* __restrict__ bconv, float* __restrict__ s,
                       float* __restrict__ t, int c) {
  int i = blockIdx.x * 128 + threadIdx.x;
  if (i < c) {
    float sc = g[i] * rsqrtf(var[i] + 1e-5f);
    s[i] = sc;
    t[i] = bconv[i] * sc + be[i] - mu[i] * sc;
  }
}

// ---------------- generic conv1 (channel GEMM) with WMMA ------------------------
// C[M,N] = W[M,K] * X[K,N] (+epilogue).  Block tile 64(M) x 128(N), K-step 32.
// MODE 0: out_f16 = acc + bias[m]
// MODE 1: out_f16 = relu(acc * s[m] + t[m])
// MODE 2: out_f32 = acc + bias[m]
template <int MODE>
__global__ __launch_bounds__(256) void gemm_ck(const _Float16* __restrict__ W,
                                               const _Float16* __restrict__ X,
                                               void* __restrict__ outp,
                                               const float* __restrict__ p0,
                                               const float* __restrict__ p1,
                                               int M, int K, int N) {
  __shared__ __align__(16) _Float16 As[64 * 32];
  __shared__ __align__(16) _Float16 Xs[32 * 128];
  const int b = blockIdx.z;
  X += (size_t)b * K * N;
  const int n0 = blockIdx.x * 128;
  const int m0 = blockIdx.y * 64;
  const int t = threadIdx.x;
  const int w = t >> 5, lane = t & 31, hi = lane >> 4;
  const int wm = w & 3, wn = w >> 2;

  v8f acc[4];
  v8f zero = {};
#pragma unroll
  for (int i = 0; i < 4; ++i) acc[i] = zero;

  for (int k0 = 0; k0 < K; k0 += 32) {
    {  // stage A tile 64x32: 256 threads x 8 halves
      int row = t >> 2, col = (t & 3) * 8;
      stage16(&As[row * 32 + col], &W[(size_t)(m0 + row) * K + k0 + col]);
    }
    {  // stage X tile 32x128: 256 threads x 16 halves
      int row = t >> 3, col = (t & 7) * 16;
      stage16(&Xs[row * 128 + col], &X[(size_t)(k0 + row) * N + n0 + col]);
      stage16(&Xs[row * 128 + col + 8], &X[(size_t)(k0 + row) * N + n0 + col + 8]);
      if (k0 + 32 < K)  // hint next tile into cache
        __builtin_prefetch(&X[(size_t)(k0 + 32 + row) * N + n0 + col], 0, 0);
    }
    stage_wait();
    __syncthreads();
    v16h a = load_a_rowmajor(As, 32, wm * 16, 0);
#pragma unroll
    for (int nf = 0; nf < 4; ++nf) {
      v16h bm = load_b_rowmajor(Xs, 128, 0, wn * 64 + nf * 16);
      acc[nf] = wmma_f16(a, bm, acc[nf]);
    }
    __syncthreads();
  }

#pragma unroll
  for (int nf = 0; nf < 4; ++nf) {
    int n = n0 + wn * 64 + nf * 16 + (lane & 15);
#pragma unroll
    for (int vv = 0; vv < 8; ++vv) {
      int m = m0 + wm * 16 + vv + 8 * hi;
      float val = acc[nf][vv];
      size_t gi = (size_t)b * M * N + (size_t)m * N + n;
      if (MODE == 0) {
        val += p0[m];
        ((_Float16*)outp)[gi] = (_Float16)val;
      } else if (MODE == 1) {
        val = fmaxf(val * p0[m] + p1[m], 0.f);
        ((_Float16*)outp)[gi] = (_Float16)val;
      } else {
        val += p0[m];
        ((float*)outp)[gi] = val;
      }
    }
  }
}

// ---------------- flash attention (per b, head, 128-query tile) -----------------
// Q,K,V are f16 [b][4 heads][64 ch][4096].  Output fused: xh = f16(feat - message).
__global__ __launch_bounds__(256) void flash_attn(const _Float16* __restrict__ Qh,
                                                  const _Float16* __restrict__ Kh,
                                                  const _Float16* __restrict__ Vh,
                                                  const float* __restrict__ feat,
                                                  _Float16* __restrict__ xh) {
  extern __shared__ __align__(16) char smem[];
  _Float16* Qs = (_Float16*)smem;     // [64][128]
  _Float16* Ks = Qs + 64 * 128;       // [64][128]
  _Float16* Vs = Ks + 64 * 128;       // [64][128]
  _Float16* Ps = Vs + 64 * 128;       // [128][128]
  float* rowm = (float*)(Ps + 128 * 128);
  float* rowl = rowm + 128;
  float* rowscale = rowl + 128;

  const int o0 = blockIdx.x * 128;
  const int h = blockIdx.y;
  const int b = blockIdx.z;
  const int t = threadIdx.x, w = t >> 5, lane = t & 31, hi = lane >> 4;
  const int wm = w & 3, wn = w >> 2;  // phase-2 ownership: 16(c) x 64(o)
  const int ow0 = w * 16;             // phase-1 ownership: 16 softmax rows

  const size_t headbase = ((size_t)b * 4 + h) * 64 * 4096;
  const _Float16* Qg = Qh + headbase;
  const _Float16* Kg = Kh + headbase;
  const _Float16* Vg = Vh + headbase;

// stage Q tile [64 c][128 o]
#pragma unroll
  for (int j = 0; j < 4; ++j) {
    int cid = t + 256 * j;
    int row = cid >> 4, col = (cid & 15) * 8;
    stage16(&Qs[row * 128 + col], &Qg[(size_t)row * 4096 + o0 + col]);
  }
  if (t < 128) {
    rowm[t] = -1e30f;
    rowl[t] = 0.f;
  }

  v8f O[4];
  v8f zero = {};
#pragma unroll
  for (int i = 0; i < 4; ++i) O[i] = zero;

  for (int i0 = 0; i0 < 4096; i0 += 128) {
    __syncthreads();  // prev phase-2 done; Qs/rowm visible on first iter
// stage K,V tiles [64 c][128 i]
#pragma unroll
    for (int j = 0; j < 4; ++j) {
      int cid = t + 256 * j;
      int row = cid >> 4, col = (cid & 15) * 8;
      stage16(&Ks[row * 128 + col], &Kg[(size_t)row * 4096 + i0 + col]);
      stage16(&Vs[row * 128 + col], &Vg[(size_t)row * 4096 + i0 + col]);
    }
    stage_wait();
    __syncthreads();

    // phase 1: S[16 o rows per wave][128 i] = (Q^T K) * 1/sqrt(64)
    v8f S[8];
#pragma unroll
    for (int fi = 0; fi < 8; ++fi) S[fi] = zero;
#pragma unroll
    for (int kk = 0; kk < 2; ++kk) {
      v16h a = load_a_colmajor(Qs, 128, ow0, kk * 32);  // A[m=o,k=c] = Qs[c][o]
#pragma unroll
      for (int fi = 0; fi < 8; ++fi) {
        v16h bm = load_b_rowmajor(Ks, 128, kk * 32, fi * 16);
        S[fi] = wmma_f16(a, bm, S[fi]);
      }
    }
#pragma unroll
    for (int fi = 0; fi < 8; ++fi)
#pragma unroll
      for (int v = 0; v < 8; ++v) S[fi][v] *= 0.125f;

    // per-row max across the 128 keys (across 8 frags, then across 16 lanes)
    float vmax[8];
#pragma unroll
    for (int v = 0; v < 8; ++v) {
      float x = S[0][v];
#pragma unroll
      for (int fi = 1; fi < 8; ++fi) x = fmaxf(x, S[fi][v]);
      vmax[v] = x;
    }
#pragma unroll
    for (int off = 1; off < 16; off <<= 1)
#pragma unroll
      for (int v = 0; v < 8; ++v) vmax[v] = fmaxf(vmax[v], __shfl_xor(vmax[v], off, 32));

    if ((lane & 15) == 0) {
#pragma unroll
      for (int v = 0; v < 8; ++v) {
        int o = ow0 + 8 * hi + v;
        float mold = rowm[o];
        float mnew = fmaxf(mold, vmax[v]);
        rowm[o] = mnew;
        float sc = __expf(mold - mnew);
        rowscale[o] = sc;
        rowl[o] *= sc;
      }
    }
    __syncthreads();

    // P = exp(S - mnew) -> Ps (f16), accumulate row sums into rowl
    float vsum[8];
#pragma unroll
    for (int v = 0; v < 8; ++v) {
      int o = ow0 + 8 * hi + v;
      float mrow = rowm[o];
      float s = 0.f;
#pragma unroll
      for (int fi = 0; fi < 8; ++fi) {
        float p = __expf(S[fi][v] - mrow);
        s += p;
        Ps[o * 128 + fi * 16 + (lane & 15)] = (_Float16)p;
      }
      vsum[v] = s;
    }
#pragma unroll
    for (int off = 1; off < 16; off <<= 1)
#pragma unroll
      for (int v = 0; v < 8; ++v) vsum[v] += __shfl_xor(vsum[v], off, 32);
    if ((lane & 15) == 0) {
#pragma unroll
      for (int v = 0; v < 8; ++v) rowl[ow0 + 8 * hi + v] += vsum[v];
    }
    __syncthreads();

    // phase 2: O = O * rowscale[o] + V * P^T
#pragma unroll
    for (int nf = 0; nf < 4; ++nf) {
      int o = wn * 64 + nf * 16 + (lane & 15);
      float sc = rowscale[o];
#pragma unroll
      for (int v = 0; v < 8; ++v) O[nf][v] *= sc;
    }
#pragma unroll
    for (int ki = 0; ki < 4; ++ki) {
      v16h a = load_a_rowmajor(Vs, 128, wm * 16, ki * 32);  // A[m=c,k=i] = Vs[c][i]
#pragma unroll
      for (int nf = 0; nf < 4; ++nf) {
        v16h bm = load_b_trans(Ps, 128, ki * 32, wn * 64 + nf * 16);  // B[k=i,n=o]=Ps[o][i]
        O[nf] = wmma_f16(a, bm, O[nf]);
      }
    }
  }
  __syncthreads();

  // epilogue: message = O / l ; xh = f16(feat - message)
#pragma unroll
  for (int nf = 0; nf < 4; ++nf) {
    int ol = wn * 64 + nf * 16 + (lane & 15);
    float inv = 1.f / rowl[ol];
#pragma unroll
    for (int v = 0; v < 8; ++v) {
      int cl = wm * 16 + v + 8 * hi;
      size_t gi = ((size_t)b * 256 + h * 64 + cl) * 4096 + o0 + ol;
      float msgv = O[nf][v] * inv;
      xh[gi] = (_Float16)(feat[gi] - msgv);
    }
  }
}

// ---------------- SE gate + final ------------------------------------------------

__global__ void colmean(const float* __restrict__ msg, float* __restrict__ sq) {
  __shared__ float red[256];
  int c = blockIdx.x, b = blockIdx.y;
  const float* p = msg + ((size_t)b * 256 + c) * 4096;
  float s = 0.f;
  for (int i = threadIdx.x; i < 4096; i += 256) s += p[i];
  red[threadIdx.x] = s;
  __syncthreads();
  for (int st = 128; st > 0; st >>= 1) {
    if (threadIdx.x < st) red[threadIdx.x] += red[threadIdx.x + st];
    __syncthreads();
  }
  if (threadIdx.x == 0) sq[b * 256 + c] = red[0] * (1.f / 4096.f);
}

__global__ void se_gate(const float* __restrict__ sq, const float* __restrict__ Wse1,
                        const float* __restrict__ bse1, const float* __restrict__ Wse2,
                        const float* __restrict__ bse2, float* __restrict__ gate) {
  __shared__ float fc[16];
  __shared__ float sv[256];
  int b = blockIdx.x, t = threadIdx.x;
  sv[t] = sq[b * 256 + t];
  __syncthreads();
  if (t < 16) {
    float a = bse1[t];
    for (int c = 0; c < 256; ++c) a += Wse1[t * 256 + c] * sv[c];
    fc[t] = fmaxf(a, 0.f);
  }
  __syncthreads();
  float g = bse2[t];
  for (int r = 0; r < 16; ++r) g += Wse2[t * 16 + r] * fc[r];
  gate[b * 256 + t] = 1.f / (1.f + __expf(-g));
}

__global__ void final_add(const float* __restrict__ feat, const float* __restrict__ msg,
                          const float* __restrict__ gate, float* __restrict__ out) {
  size_t i = (size_t)blockIdx.x * 256 + threadIdx.x;
  int bc = (int)(i >> 12);  // b*256 + c  (n = 4096)
  out[i] = feat[i] + msg[i] * gate[bc];
}

// ---------------- host orchestration ---------------------------------------------

extern "C" void kernel_launch(void* const* d_in, const int* in_sizes, int n_in,
                              void* d_out, int out_size, void* d_ws, size_t ws_size,
                              hipStream_t stream) {
  const float* feat = (const float*)d_in[0];
  const float* Wq = (const float*)d_in[1];  const float* bq = (const float*)d_in[2];
  const float* Wk = (const float*)d_in[3];  const float* bk = (const float*)d_in[4];
  const float* Wv = (const float*)d_in[5];  const float* bv = (const float*)d_in[6];
  const float* W1 = (const float*)d_in[7];  const float* b1 = (const float*)d_in[8];
  const float* g1 = (const float*)d_in[9];  const float* be1 = (const float*)d_in[10];
  const float* m1 = (const float*)d_in[11]; const float* v1 = (const float*)d_in[12];
  const float* W2 = (const float*)d_in[13]; const float* b2 = (const float*)d_in[14];
  const float* g2 = (const float*)d_in[15]; const float* be2 = (const float*)d_in[16];
  const float* m2 = (const float*)d_in[17]; const float* v2 = (const float*)d_in[18];
  const float* W3 = (const float*)d_in[19]; const float* b3 = (const float*)d_in[20];
  const float* Wse1 = (const float*)d_in[21]; const float* bse1 = (const float*)d_in[22];
  const float* Wse2 = (const float*)d_in[23]; const float* bse2 = (const float*)d_in[24];
  float* out = (float*)d_out;

  const int bs = 2, C = 256, Ch = 128, n = 4096;
  const size_t featN = (size_t)bs * C * n;   // 2M
  const size_t hN = (size_t)bs * Ch * n;     // 1M

  char* ws = (char*)d_ws;
  auto alloc = [&](size_t bytes) -> char* {
    char* p = ws;
    ws += (bytes + 255) & ~(size_t)255;
    return p;
  };
  _Float16* feat_h = (_Float16*)alloc(featN * 2);
  _Float16* Wq_h = (_Float16*)alloc((size_t)C * C * 2);
  _Float16* Wk_h = (_Float16*)alloc((size_t)C * C * 2);
  _Float16* Wv_h = (_Float16*)alloc((size_t)C * C * 2);
  _Float16* W1_h = (_Float16*)alloc((size_t)Ch * C * 2);
  _Float16* W2_h = (_Float16*)alloc((size_t)Ch * Ch * 2);
  _Float16* W3_h = (_Float16*)alloc((size_t)C * Ch * 2);
  _Float16* Qbuf = (_Float16*)alloc(featN * 2);
  _Float16* Kbuf = (_Float16*)alloc(featN * 2);
  _Float16* Vbuf = (_Float16*)alloc(featN * 2);
  _Float16* xhb  = (_Float16*)alloc(featN * 2);
  _Float16* h1b  = (_Float16*)alloc(hN * 2);
  _Float16* h2b  = (_Float16*)alloc(hN * 2);
  float* msg  = (float*)alloc(featN * 4);
  float* s1v  = (float*)alloc(Ch * 4);
  float* t1v  = (float*)alloc(Ch * 4);
  float* s2v  = (float*)alloc(Ch * 4);
  float* t2v  = (float*)alloc(Ch * 4);
  float* sqv  = (float*)alloc((size_t)bs * C * 4);
  float* gatev = (float*)alloc((size_t)bs * C * 4);

  auto cvt = [&](const float* s, _Float16* d, int cnt) {
    cvt_f32_f16<<<(cnt + 255) / 256, 256, 0, stream>>>(s, d, cnt);
  };
  cvt(feat, feat_h, (int)featN);
  cvt(Wq, Wq_h, C * C);
  cvt(Wk, Wk_h, C * C);
  cvt(Wv, Wv_h, C * C);
  cvt(W1, W1_h, Ch * C);
  cvt(W2, W2_h, Ch * Ch);
  cvt(W3, W3_h, C * Ch);
  bnprep<<<1, 128, 0, stream>>>(g1, be1, m1, v1, b1, s1v, t1v, Ch);
  bnprep<<<1, 128, 0, stream>>>(g2, be2, m2, v2, b2, s2v, t2v, Ch);

  // QKV projections (M=256,K=256,N=4096 per batch)
  dim3 gQKV(n / 128, C / 64, bs);
  gemm_ck<0><<<gQKV, 256, 0, stream>>>(Wq_h, feat_h, Qbuf, bq, nullptr, C, C, n);
  gemm_ck<0><<<gQKV, 256, 0, stream>>>(Wk_h, feat_h, Kbuf, bk, nullptr, C, C, n);
  gemm_ck<0><<<gQKV, 256, 0, stream>>>(Wv_h, feat_h, Vbuf, bv, nullptr, C, C, n);

  // flash attention + fused x = feat - message
  size_t shmem = (size_t)(3 * 64 * 128 + 128 * 128) * sizeof(_Float16) + 3 * 128 * sizeof(float);
  flash_attn<<<dim3(n / 128, 4, bs), 256, shmem, stream>>>(Qbuf, Kbuf, Vbuf, feat, xhb);

  // fc_message MLP
  gemm_ck<1><<<dim3(n / 128, Ch / 64, bs), 256, 0, stream>>>(W1_h, xhb, h1b, s1v, t1v, Ch, C, n);
  gemm_ck<1><<<dim3(n / 128, Ch / 64, bs), 256, 0, stream>>>(W2_h, h1b, h2b, s2v, t2v, Ch, Ch, n);
  gemm_ck<2><<<dim3(n / 128, C / 64, bs), 256, 0, stream>>>(W3_h, h2b, msg, b3, nullptr, C, Ch, n);

  // SE gate + residual
  colmean<<<dim3(C, bs), 256, 0, stream>>>(msg, sqv);
  se_gate<<<bs, 256, 0, stream>>>(sqv, Wse1, bse1, Wse2, bse2, gatev);
  final_add<<<(unsigned)(featN / 256), 256, 0, stream>>>(feat, msg, gatev, out);
}
